// TrajectoryGenerator_35579509080165
// MI455X (gfx1250) — compile-verified, matrix operand order checked by reading, not run on hardware
//
#include <hip/hip_runtime.h>

// ---------------- problem constants ----------------
#define HD    128       // hidden
#define GH    512       // 4*H (gates)
#define TOBS  20
#define NB    16384
#define NPRED 30
#define BM    32        // batch rows per block
#define NTH   256       // 8 waves

typedef __attribute__((ext_vector_type(16))) __bf16 bhalf16;
typedef __attribute__((ext_vector_type(8)))  float  floatx8;
typedef unsigned int u32x4 __attribute__((ext_vector_type(4)));
typedef int          i32x4 __attribute__((ext_vector_type(4)));
typedef int          i32x8 __attribute__((ext_vector_type(8)));

// ---------------- fast activations on HW transcendentals ----------------
__device__ __forceinline__ float sigm(float x) {
    // 1/(1+e^-x) = rcp(1 + 2^(-x*log2(e)))  -> v_exp_f32 + v_rcp_f32
    return __builtin_amdgcn_rcpf(1.0f + __builtin_amdgcn_exp2f(-1.4426950408889634f * x));
}
__device__ __forceinline__ float ftanh(float x) {
#if __has_builtin(__builtin_amdgcn_tanhf)
    return __builtin_amdgcn_tanhf(x);          // gfx1250 V_TANH_F32
#else
    float e = __builtin_amdgcn_exp2f(2.8853900817779268f * x);  // 2^(2x*log2e)
    return 1.0f - 2.0f * __builtin_amdgcn_rcpf(e + 1.0f);
#endif
}

// ---------------- shared memory layout (bytes) ----------------
#define OFF_SHB   0          // 512*128 bf16 packed weights      = 131072
#define OFF_SHA0  131072     // h fragments buf0 (4096 bf16)     = 8192
#define OFF_SHA1  139264     // h fragments buf1                 = 8192
#define OFF_SHHP  147456     // LN(h_pos)   fp32 [32][128]       = 16384
#define OFF_SHCP  163840     // LN(c_pos)                        = 16384
#define OFF_SHHA  180224     // LN(h_pos)+LN(h_speed)            = 16384
#define OFF_SHCA  196608     // LN(c_pos)+LN(c_speed)            = 16384
#define OFF_SHHL  212992     // fp32 h row buffer (FC / LN src)  = 16384
#define OFF_SHCL  229376     // fp32 c row buffer (LN src)       = 16384
#define OFF_SHX   245760     // x[32][2] fp32                    = 256
#define OFF_SHWIH 246016     // Wih [512][2] fp32                = 4096
#define OFF_SHBS  250112     // bih+bhh [512] fp32               = 2048
#define OFF_SHWFC 252160     // Wfc [2][128] + bfc[2] fp32       = 1032
#define SMEM_BYTES 253192

// ================= weight pre-pack: fp32 Whh[512,128] -> bf16 B-fragments ===========
__global__ __launch_bounds__(NTH) void prep_weights_kernel(
    const float* __restrict__ W0, const float* __restrict__ W1,
    const float* __restrict__ W2, const float* __restrict__ W3,
    __bf16* __restrict__ out)
{
    const float* Ws[4] = {W0, W1, W2, W3};
    unsigned flat = blockIdx.x * NTH + threadIdx.x;     // 0 .. 4*65536-1
    unsigned mat  = flat >> 16;
    unsigned idx  = flat & 65535u;
    unsigned e    = idx & 15u;
    unsigned lane = (idx >> 4) & 31u;
    unsigned kb   = (idx >> 9) & 3u;
    unsigned nt   = idx >> 11;
    unsigned n = nt * 16 + (lane & 15u);
    unsigned k = kb * 32 + ((lane >= 16) ? 8u : 0u) + ((e < 8) ? e : (e + 8));
    out[mat * 65536u + idx] = (__bf16)Ws[mat][n * HD + k];
}

// ================= TDM: 128KB weight matrix global -> LDS ==================
#define HAVE_TDM __has_builtin(__builtin_amdgcn_tensor_load_to_lds)

__device__ __forceinline__ void tdm_load_weights(const __bf16* gB, __bf16* shB)
{
#if HAVE_TDM
    unsigned long long ga = (unsigned long long)(const void*)gB;
    unsigned lds = (unsigned)(unsigned long long)(void*)shB;   // LDS byte address
    // D# group 0: count=1 user descriptor; lds_addr; 57b global addr; type=2
    u32x4 g0;
    g0[0] = 1u;
    g0[1] = lds;
    g0[2] = (unsigned)(ga & 0xffffffffu);
    g0[3] = (unsigned)((ga >> 32) & 0x01ffffffu) | 0x80000000u;
    // D# group 1: data_size=3 (8B); 1-D tile: tile_dim0 = tensor_dim0 = 16384
    // elements of 8B = 131072 bytes; tensor_dim0_stride = 16384.
    i32x8 g1;
    g1[0] = 0x00030000;      // wg_mask=0 | data_size=3<<16
    g1[1] = 0x40000000;      // tensor_dim0 low16 (16384) in [31:16]
    g1[2] = 0x00010000;      // tensor_dim0 hi16=0 | tensor_dim1=1 low16
    g1[3] = 0x40000000;      // tensor_dim1 hi16=0 | tile_dim0=16384
    g1[4] = 0;               // tile_dim1=0 (unused) | tile_dim2=0
    g1[5] = 16384;           // tensor_dim0_stride[31:0]
    g1[6] = 0;               // stride hi | tensor_dim1_stride lo
    g1[7] = 0;
    i32x4 g2 = {0, 0, 0, 0}, g3 = {0, 0, 0, 0};
#if __clang_major__ >= 23
    i32x8 g4 = {0, 0, 0, 0, 0, 0, 0, 0};
    __builtin_amdgcn_tensor_load_to_lds(g0, g1, g2, g3, g4, 0);
#else
    __builtin_amdgcn_tensor_load_to_lds(g0, g1, g2, g3, 0);
#endif
#endif
}

// ================= layernorm of one fp32 row of length 128 ==================
__device__ __forceinline__ void ln_row(const float* __restrict__ r,
                                       const float* __restrict__ g,
                                       const float* __restrict__ b,
                                       float* dst2, float* dstA, bool add)
{
    float m = 0.f;
    for (int k = 0; k < HD; ++k) m += r[k];
    m *= (1.0f / HD);
    float v = 0.f;
    for (int k = 0; k < HD; ++k) { float d = r[k] - m; v += d * d; }
    v *= (1.0f / HD);
    float inv = __builtin_amdgcn_rsqf(v + 1e-5f);
    for (int k = 0; k < HD; ++k) {
        float y = (r[k] - m) * inv * g[k] + b[k];
        if (dst2) dst2[k] = y;
        if (add) dstA[k] += y; else dstA[k] = y;
    }
}

// ================= one LSTM phase (encoder or decoder chain) =================
__device__ __forceinline__ void lstm_phase(
    const __bf16* __restrict__ gB,          // packed bf16 Whh fragments (65536)
    const float*  __restrict__ gWih,        // [512,2]
    const float*  __restrict__ gbih, const float* __restrict__ gbhh,
    const float*  __restrict__ seq,         // encoder: [T,NB,2]; decoder: nullptr
    const float*  __restrict__ x0,          // decoder: first input rows [NB,2]
    int nsteps, bool decoder,
    const float*  __restrict__ initH,       // LDS fp32 [BM,HD] or nullptr (zeros)
    const float*  __restrict__ initC,
    float* __restrict__ outBase,            // decoder: global output base
    __bf16* shB, __bf16* shA0, __bf16* shA1,
    float* shHL, float* shCL, float* shX, float* shWih, float* shBs, float* shWfc,
    int tid, int w, int lane, int lhi, int j, int row0,
    int scat0, int scat1)
{
    // ---- stage weights into LDS: Tensor Data Mover (wave 0 issues; TENSORcnt) ----
#if HAVE_TDM
    if (w == 0) {
        tdm_load_weights(gB, shB);
        __builtin_amdgcn_s_wait_tensorcnt(0);
    }
#else
    {
        const uint4* src = (const uint4*)gB;
        uint4* dst = (uint4*)shB;
        for (int i = tid; i < 8192; i += NTH) dst[i] = src[i];   // 128KB bf16
    }
#endif
    for (int i = tid; i < 2 * GH; i += NTH) shWih[i] = gWih[i];
    for (int i = tid; i < GH; i += NTH)     shBs[i]  = gbih[i] + gbhh[i];

    // ---- init state ----
    float cfrag[2][8];
    if (initH) {
#pragma unroll
        for (int mt = 0; mt < 2; ++mt)
#pragma unroll
            for (int v = 0; v < 8; ++v) {
                int m = mt * 16 + v + 8 * lhi;
                cfrag[mt][v] = initC[m * HD + j];
                int off = (mt ? scat1 : scat0) + (v + 8 * lhi) * 16;
                shA0[off] = (__bf16)initH[m * HD + j];
            }
    } else {
#pragma unroll
        for (int mt = 0; mt < 2; ++mt)
#pragma unroll
            for (int v = 0; v < 8; ++v) cfrag[mt][v] = 0.f;
        for (int i = tid; i < 512; i += NTH) ((uint4*)shA0)[i] = make_uint4(0, 0, 0, 0);
    }
    if (x0 && tid < BM) {
        float2 xv = ((const float2*)x0)[row0 + tid];
        shX[2 * tid] = xv.x; shX[2 * tid + 1] = xv.y;
    }
    __syncthreads();

    // per-lane gate constants (fixed across timesteps): n = g*128 + j
    float wn0[4], wn1[4], bsn[4];
#pragma unroll
    for (int g = 0; g < 4; ++g) {
        int n = g * HD + j;
        wn0[g] = shWih[2 * n]; wn1[g] = shWih[2 * n + 1]; bsn[g] = shBs[n];
    }

    for (int t = 0; t < nsteps; ++t) {
        __bf16* cur = (t & 1) ? shA1 : shA0;
        __bf16* nxt = (t & 1) ? shA0 : shA1;

        if (seq) {
            if (tid < BM) {
                float2 xv = ((const float2*)seq)[(size_t)t * NB + row0 + tid];
                shX[2 * tid] = xv.x; shX[2 * tid + 1] = xv.y;
            }
            if (t + 1 < nsteps)
                __builtin_prefetch(seq + ((size_t)(t + 1) * NB + row0) * 2, 0, 1);
            __syncthreads();
        }

        // ---- gates[32,512] tile: 2 M-tiles x 4 gate-tiles x 4 K-steps of WMMA ----
        floatx8 acc[4][2];
#pragma unroll
        for (int mt = 0; mt < 2; ++mt) {
            bhalf16 a[4];
#pragma unroll
            for (int kb = 0; kb < 4; ++kb)
                a[kb] = *(const bhalf16*)(cur + ((mt * 4 + kb) * 32 + lane) * 16);
#pragma unroll
            for (int g = 0; g < 4; ++g) {
                floatx8 acc8 = {0.f, 0.f, 0.f, 0.f, 0.f, 0.f, 0.f, 0.f};
#pragma unroll
                for (int kb = 0; kb < 4; ++kb) {
                    bhalf16 b = *(const bhalf16*)(shB + (((g * 8 + w) * 4 + kb) * 32 + lane) * 16);
                    acc8 = __builtin_amdgcn_wmma_f32_16x16x32_bf16(
                        false, a[kb], false, b, (short)0, acc8, false, false);
                }
                acc[g][mt] = acc8;
            }
        }

        // ---- x@Wih + biases + activations; update c; scatter h(bf16) ----
        bool lastcap = (!decoder) && (t == nsteps - 1);
#pragma unroll
        for (int mt = 0; mt < 2; ++mt) {
#pragma unroll
            for (int v = 0; v < 8; ++v) {
                int m = mt * 16 + v + 8 * lhi;
                float xv0 = shX[2 * m], xv1 = shX[2 * m + 1];
                float gi = acc[0][mt][v] + xv0 * wn0[0] + xv1 * wn1[0] + bsn[0];
                float gf = acc[1][mt][v] + xv0 * wn0[1] + xv1 * wn1[1] + bsn[1];
                float gg = acc[2][mt][v] + xv0 * wn0[2] + xv1 * wn1[2] + bsn[2];
                float go = acc[3][mt][v] + xv0 * wn0[3] + xv1 * wn1[3] + bsn[3];
                float cn = sigm(gf) * cfrag[mt][v] + sigm(gi) * ftanh(gg);
                float hn = sigm(go) * ftanh(cn);
                cfrag[mt][v] = cn;
                int off = (mt ? scat1 : scat0) + (v + 8 * lhi) * 16;
                nxt[off] = (__bf16)hn;
                if (decoder || lastcap) shHL[m * HD + j] = hn;
                if (lastcap)            shCL[m * HD + j] = cn;
            }
        }

        if (decoder) {
            __syncthreads();
            // FC: next_x[m,d] = h[m,:] @ Wfc[d,:] + bfc[d]; also emit to global out
            if (tid < 2 * BM) {
                int m = tid >> 1, d = tid & 1;
                float a = shWfc[256 + d];
#pragma unroll 8
                for (int k = 0; k < HD; ++k) a += shHL[m * HD + k] * shWfc[d * HD + k];
                outBase[((size_t)t * NB + row0 + m) * 2 + d] = a;
                shX[2 * m + d] = a;
            }
        }
        __syncthreads();
    }
}

// ================= fused trajectory kernel =================
__global__ __launch_bounds__(NTH, 1) void traj_lstm_kernel(
    const float* __restrict__ pos, const float* __restrict__ speed,
    const float* __restrict__ Wih_pe, const float* __restrict__ bih_pe, const float* __restrict__ bhh_pe,
    const float* __restrict__ Wih_se, const float* __restrict__ bih_se, const float* __restrict__ bhh_se,
    const float* __restrict__ ln_g, const float* __restrict__ ln_b,
    const float* __restrict__ Wih_sd, const float* __restrict__ bih_sd, const float* __restrict__ bhh_sd,
    const float* __restrict__ Wih_pd, const float* __restrict__ bih_pd, const float* __restrict__ bhh_pd,
    const float* __restrict__ Wfc, const float* __restrict__ bfc,
    const __bf16* __restrict__ pw,   // 4 packed Whh matrices, 65536 bf16 each
    float* __restrict__ out)         // [pos_out | speed_out], each PRED*NB*2 fp32
{
    extern __shared__ char smem[];
    __bf16* shB  = (__bf16*)(smem + OFF_SHB);
    __bf16* shA0 = (__bf16*)(smem + OFF_SHA0);
    __bf16* shA1 = (__bf16*)(smem + OFF_SHA1);
    float* shHp  = (float*)(smem + OFF_SHHP);
    float* shCp  = (float*)(smem + OFF_SHCP);
    float* shHa  = (float*)(smem + OFF_SHHA);
    float* shCa  = (float*)(smem + OFF_SHCA);
    float* shHL  = (float*)(smem + OFF_SHHL);
    float* shCL  = (float*)(smem + OFF_SHCL);
    float* shX   = (float*)(smem + OFF_SHX);
    float* shWih = (float*)(smem + OFF_SHWIH);
    float* shBs  = (float*)(smem + OFF_SHBS);
    float* shWfc = (float*)(smem + OFF_SHWFC);

    const int tid  = threadIdx.x;
    const int w    = tid >> 5;
    const int lane = tid & 31;
    const int ln   = lane & 15;
    const int lhi  = lane >> 4;
    const int j    = 16 * w + ln;          // hidden column owned by this lane
    const int row0 = blockIdx.x * BM;      // batch tile

    // stage FC weights once
    for (int i = tid; i < 256; i += NTH) shWfc[i] = Wfc[i];
    if (tid == 0) { shWfc[256] = bfc[0]; shWfc[257] = bfc[1]; }

    // precompute A-fragment scatter position for column k=j (h writeback)
    int kbj = j >> 5, kk = j & 31, ls, ee;
    if      (kk < 8)  { ls = 0; ee = kk; }
    else if (kk < 16) { ls = 1; ee = kk - 8; }
    else if (kk < 24) { ls = 0; ee = kk - 8; }
    else              { ls = 1; ee = kk - 16; }
    const int scat0 = ((0 * 4 + kbj) * 32 + ls * 16) * 16 + ee;
    const int scat1 = ((1 * 4 + kbj) * 32 + ls * 16) * 16 + ee;

    // ---- phase 1: position encoder ----
    lstm_phase(pw + 0 * 65536, Wih_pe, bih_pe, bhh_pe, pos, nullptr, TOBS, false,
               nullptr, nullptr, nullptr,
               shB, shA0, shA1, shHL, shCL, shX, shWih, shBs, shWfc,
               tid, w, lane, lhi, j, row0, scat0, scat1);
    if (tid < BM) {
        ln_row(shHL + tid * HD, ln_g, ln_b, shHp + tid * HD, shHa + tid * HD, false);
        ln_row(shCL + tid * HD, ln_g, ln_b, shCp + tid * HD, shCa + tid * HD, false);
    }
    __syncthreads();

    // ---- phase 2: speed encoder ----
    lstm_phase(pw + 1 * 65536, Wih_se, bih_se, bhh_se, speed, nullptr, TOBS, false,
               nullptr, nullptr, nullptr,
               shB, shA0, shA1, shHL, shCL, shX, shWih, shBs, shWfc,
               tid, w, lane, lhi, j, row0, scat0, scat1);
    if (tid < BM) {
        ln_row(shHL + tid * HD, ln_g, ln_b, nullptr, shHa + tid * HD, true);
        ln_row(shCL + tid * HD, ln_g, ln_b, nullptr, shCa + tid * HD, true);
    }
    __syncthreads();

    // ---- phase 3: speed decoder chain (hds,cds,cs) -> speed_out ----
    lstm_phase(pw + 2 * 65536, Wih_sd, bih_sd, bhh_sd, nullptr,
               speed + (size_t)(TOBS - 1) * NB * 2, NPRED, true,
               shHa, shCa, out + (size_t)NPRED * NB * 2,
               shB, shA0, shA1, shHL, shCL, shX, shWih, shBs, shWfc,
               tid, w, lane, lhi, j, row0, scat0, scat1);

    // ---- phase 4: position decoder chain (hpo,cpo,cp) -> pos_out ----
    lstm_phase(pw + 3 * 65536, Wih_pd, bih_pd, bhh_pd, nullptr,
               pos + (size_t)(TOBS - 1) * NB * 2, NPRED, true,
               shHp, shCp, out,
               shB, shA0, shA1, shHL, shCL, shX, shWih, shBs, shWfc,
               tid, w, lane, lhi, j, row0, scat0, scat1);
}

// ================= launch =================
extern "C" void kernel_launch(void* const* d_in, const int* in_sizes, int n_in,
                              void* d_out, int out_size, void* d_ws, size_t ws_size,
                              hipStream_t stream)
{
    const float* pos    = (const float*)d_in[0];
    const float* speed  = (const float*)d_in[1];
    const float* Wih_pe = (const float*)d_in[2];
    const float* Whh_pe = (const float*)d_in[3];
    const float* bih_pe = (const float*)d_in[4];
    const float* bhh_pe = (const float*)d_in[5];
    const float* Wih_se = (const float*)d_in[6];
    const float* Whh_se = (const float*)d_in[7];
    const float* bih_se = (const float*)d_in[8];
    const float* bhh_se = (const float*)d_in[9];
    const float* ln_g   = (const float*)d_in[10];
    const float* ln_b   = (const float*)d_in[11];
    const float* Wih_sd = (const float*)d_in[12];
    const float* Whh_sd = (const float*)d_in[13];
    const float* bih_sd = (const float*)d_in[14];
    const float* bhh_sd = (const float*)d_in[15];
    const float* Wih_pd = (const float*)d_in[16];
    const float* Whh_pd = (const float*)d_in[17];
    const float* bih_pd = (const float*)d_in[18];
    const float* bhh_pd = (const float*)d_in[19];
    const float* Wfc    = (const float*)d_in[20];
    const float* bfc    = (const float*)d_in[21];
    float* out = (float*)d_out;
    __bf16* pw = (__bf16*)d_ws;   // 4 * 65536 bf16 = 512 KB

    // one-time bf16 fragment pack of the four Whh matrices
    prep_weights_kernel<<<4 * 65536 / NTH, NTH, 0, stream>>>(
        Whh_pe, Whh_se, Whh_sd, Whh_pd, pw);

    traj_lstm_kernel<<<NB / BM, NTH, SMEM_BYTES, stream>>>(
        pos, speed,
        Wih_pe, bih_pe, bhh_pe,
        Wih_se, bih_se, bhh_se,
        ln_g, ln_b,
        Wih_sd, bih_sd, bhh_sd,
        Wih_pd, bih_pd, bhh_pd,
        Wfc, bfc, pw, out);
}
